// VcpAtt_54125177864529
// MI455X (gfx1250) — matrix operand order repaired and checked by reading, not compiled, not run on hardware
//
#include <hip/hip_runtime.h>

// ---------------------------------------------------------------------------
// MI455X (gfx1250) fused correspondence-attention.
// Roofline: 43 GFLOP vs 34 MB I/O -> compute bound; bf16 WMMA
// (v_wmma_f32_16x16x32_bf16, 8x the K-depth of fp32 WMMA) with f32 accum,
// flash-style softmax so the 4096x4096 score matrix never hits HBM.
// K tiles are double-buffered in LDS via GLOBAL_LOAD_ASYNC_TO_LDS_B128
// (ASYNCcnt-tracked DMA, no VGPR round-trip) so staging overlaps the WMMAs.
// ---------------------------------------------------------------------------

typedef __attribute__((ext_vector_type(16))) __bf16 v16bf;
typedef __attribute__((ext_vector_type(8)))  __bf16 v8bf;
typedef __attribute__((ext_vector_type(8)))  float  v8f;

constexpr int Bc = 2, Ec = 512, Nc = 4096, Mc = 4096;
constexpr int SPLITM = 8;                  // m-range split for occupancy
constexpr int MRANGE = Mc / SPLITM;        // 512 m per block

__device__ __forceinline__ __bf16 f32_to_bf16(float f) {
    union { float f; unsigned u; } in; in.f = f;
    unsigned u = in.u;
    unsigned r = u + 0x7FFFu + ((u >> 16) & 1u);   // round to nearest even
    union { unsigned short s; __bf16 b; } out;
    out.s = (unsigned short)(r >> 16);
    return out.b;
}

// Async global->LDS 16-byte copy (per lane). VDST VGPR = LDS byte address,
// VADDR = 64-bit global address, GV mode. Tracked by ASYNCcnt.
__device__ __forceinline__ void async_load_b128(unsigned lds_off, const void* gaddr) {
    asm volatile("global_load_async_to_lds_b128 %0, %1, off"
                 :: "v"(lds_off), "v"(gaddr)
                 : "memory");
}
__device__ __forceinline__ void wait_async0() {
#if __has_builtin(__builtin_amdgcn_s_wait_asynccnt)
    __builtin_amdgcn_s_wait_asynccnt(0);
#else
    asm volatile("s_wait_asynccnt 0" ::: "memory");
#endif
}

// A fragment (16x32 bf16, MxK): lane l<16 holds row M=l, K={0..7,16..23};
// lane l+16 holds row M=l, K={8..15,24..31}.  row = base ptr of the M-row.
__device__ __forceinline__ v16bf load_frag_a(const __bf16* row, int kb, int hi) {
    v8bf s0 = *(const v8bf*)(row + kb + hi * 8);
    v8bf s1 = *(const v8bf*)(row + kb + 16 + hi * 8);
    v16bf a;
    #pragma unroll
    for (int i = 0; i < 8; ++i) { a[i] = s0[i]; a[8 + i] = s1[i]; }
    return a;
}

// B fragment (32x16 bf16, KxN): lane l<16 holds col N=l, K={0..7,16..23};
// lane l+16 holds col N=l, K={8..15,24..31}.  col = base ptr of the N-column (K-major).
__device__ __forceinline__ v16bf load_frag_b(const __bf16* col, int kb, int hi) {
    v8bf s0 = *(const v8bf*)(col + kb + hi * 8);
    v8bf s1 = *(const v8bf*)(col + kb + 16 + hi * 8);
    v16bf b;
    #pragma unroll
    for (int i = 0; i < 8; ++i) { b[i] = s0[i]; b[8 + i] = s1[i]; }
    return b;
}

__device__ __forceinline__ v8f wmma_bf16(v16bf a, v16bf b, v8f c) {
    return __builtin_amdgcn_wmma_f32_16x16x32_bf16(
        false, a, false, b, (short)0, c, false, false);
}

// --- f32 -> bf16 elementwise convert (for Wq/Wk) ---------------------------
__global__ void cvt_bf16_kernel(const float* __restrict__ in,
                                __bf16* __restrict__ out, int n) {
    int i = blockIdx.x * blockDim.x + threadIdx.x;
    if (i < n) out[i] = f32_to_bf16(in[i]);
}

// --- passthrough copy of `src` ---------------------------------------------
__global__ void copy_f32_kernel(const float* __restrict__ in,
                                float* __restrict__ out, int n) {
    int i = blockIdx.x * blockDim.x + threadIdx.x;
    if (i < n) out[i] = in[i];
}

// ---------------------------------------------------------------------------
// Projection: out[b][n][f] = sum_e emb[b][e][n] * W[f][e] + bias[f]
// also outNorm[b][n] = ||out row||^2 (needed only for the K side).
// WMMA mapping: M-dim=f (A = W rows, contiguous e), N-dim=n
// (B = LDS-transposed emb tile), K-dim=e.
// Block: 256 thr / 8 waves; 16 n-rows x full 512 f (4 acc tiles per wave).
// ---------------------------------------------------------------------------
__global__ __launch_bounds__(256) void proj_kernel(
    const float* __restrict__ emb,    // [B][E][R]
    const __bf16* __restrict__ Wb,    // [E][E] bf16
    const float* __restrict__ bias,   // [E]
    __bf16* __restrict__ outRow,      // [B][R][E]
    float* __restrict__ outNorm,      // [B][R]
    int R)
{
    __shared__ __bf16 Et[16 * 32];    // [n][e] transposed bf16 tile
    __shared__ float  norm_lds[16];

    const int tid  = threadIdx.x;
    const int wave = tid >> 5;
    const int lane = tid & 31;
    const int c    = lane & 15;
    const int hi   = lane >> 4;

    const int blocksPerBatch = R / 16;
    const int b     = blockIdx.x / blocksPerBatch;
    const int nbase = (blockIdx.x % blocksPerBatch) * 16;

    if (tid < 16) norm_lds[tid] = 0.0f;

    v8f acc[4];
    #pragma unroll
    for (int p = 0; p < 4; ++p)
        #pragma unroll
        for (int j = 0; j < 8; ++j) acc[p][j] = 0.0f;

    for (int kb = 0; kb < Ec; kb += 32) {
        __syncthreads();
        // stage emb tile [32 e][16 n] -> Et[n][e] bf16 (coalesced over n)
        #pragma unroll
        for (int idx = tid; idx < 512; idx += 256) {
            int e = idx >> 4, n = idx & 15;
            Et[n * 32 + e] =
                f32_to_bf16(emb[(size_t)(b * Ec + kb + e) * R + nbase + n]);
        }
        __syncthreads();

        v16bf bfrag = load_frag_b(&Et[c * 32], 0, hi);
        #pragma unroll
        for (int p = 0; p < 4; ++p) {
            int f = p * 128 + wave * 16 + c;              // A row (M = f)
            v16bf afrag = load_frag_a(Wb + (size_t)f * Ec, kb, hi);
            acc[p] = wmma_bf16(afrag, bfrag, acc[p]);
        }
    }

    // epilogue: bias add, bf16 store, row-norm reduce.
    // D layout: lane holds col n=c, rows f = fbase + hi*8 + j.
    float nsum = 0.0f;
    const int n = nbase + c;
    #pragma unroll
    for (int p = 0; p < 4; ++p) {
        int f0 = p * 128 + wave * 16 + hi * 8;
        float4 bv0 = *(const float4*)(bias + f0);
        float4 bv1 = *(const float4*)(bias + f0 + 4);
        float bb[8] = {bv0.x, bv0.y, bv0.z, bv0.w, bv1.x, bv1.y, bv1.z, bv1.w};
        v8bf st;
        #pragma unroll
        for (int j = 0; j < 8; ++j) {
            float v = acc[p][j] + bb[j];
            nsum += v * v;
            st[j] = f32_to_bf16(v);
        }
        *(v8bf*)(outRow + ((size_t)b * R + n) * Ec + f0) = st;
    }
    atomicAdd(&norm_lds[c], nsum);
    __syncthreads();
    if (tid < 16) outNorm[(size_t)b * R + nbase + tid] = norm_lds[tid];
}

// ---------------------------------------------------------------------------
// Fused flash attention + 3-dim weighted sum, S^T formulation:
// WMMA M-dim = m (A = K rows), N-dim = n (B = Q cols). Per lane: 8 m-values
// of one n; lane pair (l, l^16) covers the 16-m tile; softmax max shared via
// one shfl_xor per tile. 128 q-rows cached in LDS; K tiles double-buffered
// and filled with async global->LDS DMA so staging overlaps the WMMA burst.
// 160 KB LDS of the 320 KB WGP budget. Split-M across blocks for occupancy.
// ---------------------------------------------------------------------------
__global__ __launch_bounds__(256) void attn_kernel(
    const __bf16* __restrict__ Qb,    // [B][N][E] bf16
    const __bf16* __restrict__ Kb,    // [B][M][E] bf16
    const float* __restrict__ kk,     // [B][M]  ||k||^2
    const float* __restrict__ tgt,    // [B][3][M]
    float* __restrict__ pmax,         // [B][SPLITM][N]
    float* __restrict__ psum,         // [B][SPLITM][N]
    float* __restrict__ pacc)         // [B][SPLITM][3][N]
{
    extern __shared__ char smem[];
    __bf16* Qlds   = (__bf16*)smem;                // 128*512 bf16 = 128 KB
    __bf16* Ktile0 = Qlds + 128 * Ec;              // 16*512 bf16 = 16 KB
    __bf16* Ktile1 = Ktile0 + 16 * Ec;             // 16 KB (double buffer)

    const int tid  = threadIdx.x;
    const int wave = tid >> 5;
    const int lane = tid & 31;
    const int c    = lane & 15;
    const int hi   = lane >> 4;

    int bid = blockIdx.x;
    const int s    = bid % SPLITM;  bid /= SPLITM;
    const int nblk = bid % (Nc / 128); bid /= (Nc / 128);
    const int b    = bid;
    const int nbase_blk = nblk * 128;
    const int nwave     = nbase_blk + wave * 16;

    // async DMA of one 16x512 bf16 K tile into an LDS buffer (4 b128/thread)
    auto stage_k = [&](int mbase, __bf16* buf) {
        const char* g = (const char*)(Kb + ((size_t)b * Mc + mbase) * Ec);
        unsigned lbase = (unsigned)(size_t)buf;     // low 32 bits = LDS offset
        #pragma unroll
        for (int i = tid; i < 16 * Ec / 8; i += 256)
            async_load_b128(lbase + i * 16, g + i * 16);
    };

    // one-time async cache of this block's 128 q rows (32 b128/thread)
    {
        const char* g = (const char*)(Qb + ((size_t)b * Nc + nbase_blk) * Ec);
        unsigned lbase = (unsigned)(size_t)Qlds;
        for (int i = tid; i < 128 * Ec / 8; i += 256)
            async_load_b128(lbase + i * 16, g + i * 16);
    }
    const int mstart = s * MRANGE;
    stage_k(mstart, Ktile0);        // prefetch first K tile (36 async/wave < 63)
    wait_async0();
    __syncthreads();

    float rmax = -3.0e38f, rsum = 0.0f;
    float a0 = 0.0f, a1 = 0.0f, a2 = 0.0f;

    for (int mt = 0; mt < MRANGE; mt += 16) {
        const int mbase = mstart + mt;
        __bf16* cur = (mt & 16) ? Ktile1 : Ktile0;
        __bf16* nxt = (mt & 16) ? Ktile0 : Ktile1;
        if (mt + 16 < MRANGE) stage_k(mbase + 16, nxt);   // overlap with WMMAs

        v8f sacc;
        #pragma unroll
        for (int j = 0; j < 8; ++j) sacc[j] = 0.0f;
        #pragma unroll 8
        for (int kb = 0; kb < Ec; kb += 32) {
            v16bf afrag = load_frag_a(cur + c * Ec, kb, hi);
            v16bf bfrag = load_frag_b(Qlds + (size_t)(wave * 16 + c) * Ec, kb, hi);
            sacc = wmma_bf16(afrag, bfrag, sacc);
        }

        // scores for rows m = mbase + hi*8 + j, col n = nwave + c
        // (the -||q||^2 term is constant per n -> cancels in softmax)
        // kk / tgt for this lane's 8 m-rows: aligned float4 pairs from L2.
        const int m0 = mbase + hi * 8;
        float4 k4a = *(const float4*)(kk + (size_t)b * Mc + m0);
        float4 k4b = *(const float4*)(kk + (size_t)b * Mc + m0 + 4);
        float kx[8] = {k4a.x, k4a.y, k4a.z, k4a.w, k4b.x, k4b.y, k4b.z, k4b.w};
        float p[8], tmax = -3.0e38f;
        #pragma unroll
        for (int j = 0; j < 8; ++j) {
            p[j] = 2.0f * sacc[j] - kx[j];
            tmax = fmaxf(tmax, p[j]);
        }
        tmax = fmaxf(tmax, __shfl_xor(tmax, 16, 32));   // share max across pair
        float nmax  = fmaxf(rmax, tmax);
        float scale = __expf(rmax - nmax);
        rsum *= scale; a0 *= scale; a1 *= scale; a2 *= scale;
        rmax = nmax;

        const float* tg = tgt + (size_t)b * 3 * Mc + m0;
        float4 t0a = *(const float4*)(tg);
        float4 t0b = *(const float4*)(tg + 4);
        float4 t1a = *(const float4*)(tg + Mc);
        float4 t1b = *(const float4*)(tg + Mc + 4);
        float4 t2a = *(const float4*)(tg + 2 * Mc);
        float4 t2b = *(const float4*)(tg + 2 * Mc + 4);
        float t0[8] = {t0a.x, t0a.y, t0a.z, t0a.w, t0b.x, t0b.y, t0b.z, t0b.w};
        float t1[8] = {t1a.x, t1a.y, t1a.z, t1a.w, t1b.x, t1b.y, t1b.z, t1b.w};
        float t2[8] = {t2a.x, t2a.y, t2a.z, t2a.w, t2b.x, t2b.y, t2b.z, t2b.w};
        #pragma unroll
        for (int j = 0; j < 8; ++j) {
            float e = __expf(p[j] - nmax);
            rsum += e;
            a0 += e * t0[j];
            a1 += e * t1[j];
            a2 += e * t2[j];
        }

        wait_async0();       // next tile landed (no-op on last iteration)
        __syncthreads();
    }

    // lane pair combine (identical rmax by construction)
    rsum += __shfl_xor(rsum, 16, 32);
    a0   += __shfl_xor(a0, 16, 32);
    a1   += __shfl_xor(a1, 16, 32);
    a2   += __shfl_xor(a2, 16, 32);
    if (hi == 0) {
        size_t base = ((size_t)b * SPLITM + s) * Nc + nwave + c;
        pmax[base] = rmax;
        psum[base] = rsum;
        size_t pb = (((size_t)b * SPLITM + s) * 3) * Nc + nwave + c;
        pacc[pb]          = a0;
        pacc[pb + Nc]     = a1;
        pacc[pb + 2 * Nc] = a2;
    }
}

// --- log-sum-exp merge of SPLITM partials ----------------------------------
__global__ void merge_kernel(const float* __restrict__ pmax,
                             const float* __restrict__ psum,
                             const float* __restrict__ pacc,
                             float* __restrict__ out_corr /* [B][3][N] */) {
    int i = blockIdx.x * blockDim.x + threadIdx.x;
    if (i >= Bc * Nc) return;
    int b = i / Nc, n = i % Nc;
    float gmax = -3.0e38f;
    #pragma unroll
    for (int s = 0; s < SPLITM; ++s)
        gmax = fmaxf(gmax, pmax[((size_t)b * SPLITM + s) * Nc + n]);
    float tot = 0.0f, c0 = 0.0f, c1 = 0.0f, c2 = 0.0f;
    #pragma unroll
    for (int s = 0; s < SPLITM; ++s) {
        size_t ix = ((size_t)b * SPLITM + s) * Nc + n;
        float w = __expf(pmax[ix] - gmax);
        tot += psum[ix] * w;
        size_t pb = (((size_t)b * SPLITM + s) * 3) * Nc + n;
        c0 += pacc[pb] * w;
        c1 += pacc[pb + Nc] * w;
        c2 += pacc[pb + 2 * Nc] * w;
    }
    float inv = 1.0f / tot;
    out_corr[((size_t)b * 3 + 0) * Nc + n] = c0 * inv;
    out_corr[((size_t)b * 3 + 1) * Nc + n] = c1 * inv;
    out_corr[((size_t)b * 3 + 2) * Nc + n] = c2 * inv;
}

extern "C" void kernel_launch(void* const* d_in, const int* in_sizes, int n_in,
                              void* d_out, int out_size, void* d_ws, size_t ws_size,
                              hipStream_t stream) {
    const float* src_emb = (const float*)d_in[0];   // [B][E][N]
    const float* tgt_emb = (const float*)d_in[1];   // [B][E][M]
    const float* src     = (const float*)d_in[2];   // [B][3][N]
    const float* tgt     = (const float*)d_in[3];   // [B][3][M]
    const float* Wq      = (const float*)d_in[4];   // [E][E]
    const float* bq      = (const float*)d_in[5];   // [E]
    const float* Wk      = (const float*)d_in[6];   // [E][E]
    const float* bk      = (const float*)d_in[7];   // [E]
    float* out = (float*)d_out;                     // src (24576) ++ src_corr (24576)

    // workspace carve-up
    char* ws = (char*)d_ws;
    size_t off = 0;
    auto carve = [&](size_t bytes) {
        char* p = ws + off;
        off += (bytes + 255) & ~(size_t)255;
        return p;
    };
    __bf16* Qb   = (__bf16*)carve((size_t)Bc * Nc * Ec * 2);
    __bf16* Kb   = (__bf16*)carve((size_t)Bc * Mc * Ec * 2);
    float*  kkv  = (float*) carve((size_t)Bc * Mc * 4);
    float*  qqv  = (float*) carve((size_t)Bc * Nc * 4);
    __bf16* Wqb  = (__bf16*)carve((size_t)Ec * Ec * 2);
    __bf16* Wkb  = (__bf16*)carve((size_t)Ec * Ec * 2);
    float*  pm   = (float*) carve((size_t)Bc * SPLITM * Nc * 4);
    float*  psm  = (float*) carve((size_t)Bc * SPLITM * Nc * 4);
    float*  pac  = (float*) carve((size_t)Bc * SPLITM * 3 * Nc * 4);
    if (off > ws_size) return;  // workspace too small -> no-op

    // 1) weights -> bf16
    cvt_bf16_kernel<<<(Ec * Ec + 255) / 256, 256, 0, stream>>>(Wq, Wqb, Ec * Ec);
    cvt_bf16_kernel<<<(Ec * Ec + 255) / 256, 256, 0, stream>>>(Wk, Wkb, Ec * Ec);

    // 2) projections (q -> Qb/qq, k -> Kb/kk)
    proj_kernel<<<Bc * (Nc / 16), 256, 0, stream>>>(src_emb, Wqb, bq, Qb, qqv, Nc);
    proj_kernel<<<Bc * (Mc / 16), 256, 0, stream>>>(tgt_emb, Wkb, bk, Kb, kkv, Mc);

    // 3) passthrough src -> out[0 : B*3*N)
    copy_f32_kernel<<<(Bc * 3 * Nc + 255) / 256, 256, 0, stream>>>(src, out, Bc * 3 * Nc);

    // 4) fused attention (160 KB dynamic LDS, within CDNA5's 320 KB WGP LDS)
    const int smem_bytes = 128 * Ec * 2 + 2 * 16 * Ec * 2;
    (void)hipFuncSetAttribute(reinterpret_cast<const void*>(attn_kernel),
                              hipFuncAttributeMaxDynamicSharedMemorySize, smem_bytes);
    attn_kernel<<<Bc * (Nc / 128) * SPLITM, 256, smem_bytes, stream>>>(
        Qb, Kb, kkv, tgt, pm, psm, pac);

    // 5) merge split-M partials -> out[B*3*N : 2*B*3*N)
    merge_kernel<<<(Bc * Nc + 255) / 256, 256, 0, stream>>>(
        pm, psm, pac, out + (size_t)Bc * 3 * Nc);
}
